// TransformerEncoderLayer_919123001548
// MI455X (gfx1250) — compile-verified, matrix operand order checked by reading, not run on hardware
//
#include <hip/hip_runtime.h>

// ---------------------------------------------------------------------------
// Types for CDNA5 WMMA (wave32): v_wmma_f32_16x16x32_bf16
// ---------------------------------------------------------------------------
typedef __attribute__((ext_vector_type(16))) __bf16 v16bf_t;
typedef __attribute__((ext_vector_type(8)))  float  v8f_t;
typedef __attribute__((ext_vector_type(4)))  unsigned int v4u_t;
typedef __attribute__((ext_vector_type(8)))  int    v8i_t;
typedef __attribute__((ext_vector_type(4)))  int    v4i_t;

union Frag16 { uint4 u[2]; v16bf_t v; };

#define TOKENS 16384      // B*S
#define DMODEL 1024
#define NHEADS 8
#define HDIM   128

#define BM 256            // block tile M
#define BN 128            // block tile N
#define BK 32             // K step (one WMMA K per stage)
#define LDSROW 40         // BK + 8 pad; 80 bytes/row -> 16B aligned, conflict-free

#if __has_builtin(__builtin_amdgcn_tensor_load_to_lds) && \
    __has_builtin(__builtin_amdgcn_s_wait_tensorcnt)
#define HAVE_TDM 1
#else
#define HAVE_TDM 0
#endif

__device__ __forceinline__ unsigned short f2bf(float f) {
  unsigned int u = __float_as_uint(f);
  u += 0x7FFFu + ((u >> 16) & 1u);          // round-to-nearest-even
  return (unsigned short)(u >> 16);
}
__device__ __forceinline__ float bf2f(unsigned short h) {
  return __uint_as_float(((unsigned int)h) << 16);
}

// Async global->LDS copy, 16 bytes/lane, tracked by ASYNCcnt (CDNA5 fallback path).
__device__ __forceinline__ void async_copy_b128(void* lds_dst, const void* gsrc) {
  unsigned lds_off = (unsigned)(size_t)lds_dst;   // low 32 bits of generic = LDS offset
  asm volatile("global_load_async_to_lds_b128 %0, %1, off"
               :: "v"(lds_off), "v"(gsrc) : "memory");
}
__device__ __forceinline__ void wait_async_all() {
  asm volatile("s_wait_asynccnt 0x0" ::: "memory");
}

#if HAVE_TDM
// ---------------------------------------------------------------------------
// TDM: one descriptor moves a [tile_rows x 32] bf16 tile (row stride 1024) into
// LDS with hardware padding: rows are 64B; pad_interval=3 (every 64B) and
// pad_amount=3 (4 dwords) give the 80B LDS row stride (LDSROW=40 bf16).
// Toolchain builtin arity: 6 args (g0 v4u, g1 v8i, g2 v4i, g3 v4i, v8i, cpol).
// ---------------------------------------------------------------------------
__device__ __forceinline__ void tdm_load_tile(void* lds_dst, const void* gsrc,
                                              unsigned tile_rows, unsigned tensor_rows) {
  unsigned lds_off = (unsigned)(size_t)lds_dst;
  unsigned long long ga = (unsigned long long)(size_t)gsrc;

  v4u_t g0;
  g0[0] = 1u;                                              // count=1, user mode
  g0[1] = lds_off;                                         // lds_addr [63:32]
  g0[2] = (unsigned)ga;                                    // global_addr lo
  g0[3] = (unsigned)((ga >> 32) & 0x01FFFFFFu) | (2u << 30); // addr hi | type=2

  v8i_t g1;
  g1[0] = (int)((1u << 16)   // data_size = 2 bytes
              | (1u << 20)   // pad_enable
              | (3u << 22)   // pad_interval: 2^3 * 8B = 64B
              | (3u << 25)); // pad_amount: 4 dwords = 16B
  g1[1] = (int)(((unsigned)DMODEL & 0xFFFFu) << 16);       // tensor_dim0 lo16
  g1[2] = (int)(((unsigned)DMODEL >> 16) |
                ((tensor_rows & 0xFFFFu) << 16));          // dim0 hi | dim1 lo
  g1[3] = (int)((tensor_rows >> 16) |
                ((unsigned)BK << 16));                     // dim1 hi | tile_dim0
  g1[4] = (int)(tile_rows & 0xFFFFu);                      // tile_dim1 | tile_dim2=0
  g1[5] = (int)DMODEL;                                     // tensor_dim0_stride lo32
  g1[6] = 0;                                               // stride0 hi | stride1 lo
  g1[7] = 0;                                               // stride1 hi

  v4i_t z4 = {0, 0, 0, 0};                                 // 2D: groups 2/3 unused
  v8i_t z8 = {0, 0, 0, 0, 0, 0, 0, 0};
  __builtin_amdgcn_tensor_load_to_lds(g0, g1, z4, z4, z8, 0);
}
#endif

// ---------------------------------------------------------------------------
// fp32 -> bf16 bulk convert (grid-stride, 4 elems/thread/iter)
// ---------------------------------------------------------------------------
__global__ __launch_bounds__(256)
void cvt_f32_bf16_kernel(const float* __restrict__ x,
                         unsigned short* __restrict__ y, long n4) {
  long i = (long)blockIdx.x * blockDim.x + threadIdx.x;
  long stride = (long)gridDim.x * blockDim.x;
  const float4* x4 = (const float4*)x;
  for (; i < n4; i += stride) {
    float4 f = x4[i];
    ushort4 r;
    r.x = f2bf(f.x); r.y = f2bf(f.y); r.z = f2bf(f.z); r.w = f2bf(f.w);
    *(ushort4*)(y + 4 * i) = r;
  }
}

// ---------------------------------------------------------------------------
// W[K=1024][N=1024] fp32  ->  Wt[N][K] bf16   (32x32 LDS tiles)
// ---------------------------------------------------------------------------
__global__ __launch_bounds__(256)
void transpose_cvt_kernel(const float* __restrict__ W,
                          unsigned short* __restrict__ Wt) {
  __shared__ float tile[32][33];
  const int bn = blockIdx.x * 32;          // n base
  const int bk = blockIdx.y * 32;          // k base
  const int tx = threadIdx.x & 31;
  const int ty = threadIdx.x >> 5;         // 0..7
  #pragma unroll
  for (int i = 0; i < 32; i += 8)
    tile[ty + i][tx] = W[(size_t)(bk + ty + i) * DMODEL + bn + tx];
  __syncthreads();
  #pragma unroll
  for (int i = 0; i < 32; i += 8)
    Wt[(size_t)(bn + ty + i) * DMODEL + bk + tx] = f2bf(tile[tx][ty + i]);
}

// ---------------------------------------------------------------------------
// C[16384][1024] = A[M][K](bf16) @ Bt[N][K](bf16)^T + bias, optional ReLU.
// Block 256x128x32, 256 threads = 8 waves (4x2); wave = 4x4 16x16 WMMA tiles.
// Staging: TDM tensor loads (TENSORcnt) if available, else async-to-LDS.
// K loop kept rolled (#pragma unroll 1) so the 128-VGPR accumulator set is
// allocated once (no cross-iteration acc copies / VGPR-MSB pressure).
// ---------------------------------------------------------------------------
__global__ __launch_bounds__(256, 1)
void gemm_bf16_wmma_kernel(const unsigned short* __restrict__ A,
                           const unsigned short* __restrict__ Bt,
                           const float* __restrict__ bias,
                           void* __restrict__ C,
                           int c_is_f32, int relu) {
  __shared__ unsigned short As[2][BM][LDSROW];   // 40960 B
  __shared__ unsigned short Bs[2][BN][LDSROW];   // 20480 B

  const int tid  = threadIdx.x;
  const int lane = tid & 31;
  const int wave = tid >> 5;       // 0..7
  const int wm   = wave >> 1;      // 0..3  (64 rows each)
  const int wn   = wave & 1;       // 0..1  (64 cols each)
  const int lr   = lane & 15;      // row/col within a 16x16 tile
  const int lh   = lane >> 4;      // K-half select (16-bit A layout, ISA 7.12.2)
  const int m0   = blockIdx.y * BM;
  const int n0   = blockIdx.x * BN;

  // accumulators (C/D layout: all 8 elements of a lane share one column)
  v8f_t acc[4][4];
  #pragma unroll
  for (int nt = 0; nt < 4; ++nt) {
    float bv = bias[n0 + wn * 64 + nt * 16 + lr];
    #pragma unroll
    for (int mt = 0; mt < 4; ++mt)
      #pragma unroll
      for (int r = 0; r < 8; ++r) acc[mt][nt][r] = bv;
  }

  auto load_stage = [&](int buf, int k0) {
#if HAVE_TDM
    if (wave == 0) {     // one DMA pair per workgroup per stage; EXEC-independent
      tdm_load_tile(&As[buf][0][0], A + (size_t)m0 * DMODEL + k0, BM, TOKENS);
      tdm_load_tile(&Bs[buf][0][0], Bt + (size_t)n0 * DMODEL + k0, BN, DMODEL);
    }
#else
    #pragma unroll
    for (int i = 0; i < 4; ++i) {
      int idx = tid + i * 256;
      int row = idx >> 2;
      int ch  = (idx & 3) * 8;
      async_copy_b128(&As[buf][row][ch],
                      A + (size_t)(m0 + row) * DMODEL + k0 + ch);
    }
    #pragma unroll
    for (int i = 0; i < 2; ++i) {
      int idx = tid + i * 256;
      int row = idx >> 2;
      int ch  = (idx & 3) * 8;
      async_copy_b128(&Bs[buf][row][ch],
                      Bt + (size_t)(n0 + row) * DMODEL + k0 + ch);
    }
#endif
  };

  auto wait_stage = [&]() {
#if HAVE_TDM
    if (wave == 0) __builtin_amdgcn_s_wait_tensorcnt(0);
#else
    wait_async_all();
#endif
    __syncthreads();
  };

  load_stage(0, 0);
  wait_stage();

  const int KT = DMODEL / BK;      // 32
  #pragma unroll 1
  for (int kt = 0; kt < KT; ++kt) {
    const int buf = kt & 1;
    if (kt + 1 < KT) load_stage(buf ^ 1, (kt + 1) * BK);

    Frag16 af[4], bfr[4];
    #pragma unroll
    for (int mt = 0; mt < 4; ++mt) {
      const unsigned short* p = &As[buf][wm * 64 + mt * 16 + lr][lh * 8];
      af[mt].u[0] = *(const uint4*)p;          // K = kb .. kb+7
      af[mt].u[1] = *(const uint4*)(p + 16);   // K = kb+16 .. kb+23
    }
    #pragma unroll
    for (int nt = 0; nt < 4; ++nt) {
      const unsigned short* p = &Bs[buf][wn * 64 + nt * 16 + lr][lh * 8];
      bfr[nt].u[0] = *(const uint4*)p;
      bfr[nt].u[1] = *(const uint4*)(p + 16);
    }

    #pragma unroll
    for (int mt = 0; mt < 4; ++mt)
      #pragma unroll
      for (int nt = 0; nt < 4; ++nt)
        acc[mt][nt] = __builtin_amdgcn_wmma_f32_16x16x32_bf16(
            false, af[mt].v, false, bfr[nt].v,
            (short)0, acc[mt][nt], false, false);

    wait_stage();        // stage kt+1 landed in LDS; safe to flip buffers
  }

  // epilogue
  #pragma unroll
  for (int mt = 0; mt < 4; ++mt) {
    #pragma unroll
    for (int nt = 0; nt < 4; ++nt) {
      const int col = n0 + wn * 64 + nt * 16 + lr;
      #pragma unroll
      for (int r = 0; r < 8; ++r) {
        const int row = m0 + wm * 64 + mt * 16 + lh * 8 + r;
        float val = acc[mt][nt][r];
        if (relu) val = fmaxf(val, 0.0f);
        if (c_is_f32)
          ((float*)C)[(size_t)row * DMODEL + col] = val;
        else
          ((unsigned short*)C)[(size_t)row * DMODEL + col] = f2bf(val);
      }
    }
  }
}

// ---------------------------------------------------------------------------
// Per-token 8x8 head attention. 1 thread per (token, head).
// ---------------------------------------------------------------------------
__global__ __launch_bounds__(256)
void attention_kernel(const unsigned short* __restrict__ q,
                      const unsigned short* __restrict__ k,
                      const unsigned short* __restrict__ v,
                      unsigned short* __restrict__ outb,
                      float* __restrict__ attnw) {
  const int gid = blockIdx.x * 256 + threadIdx.x;   // 0 .. TOKENS*8-1
  const int tok = gid >> 3;
  const int h   = gid & 7;
  const unsigned short* qp = q + (size_t)tok * DMODEL + h * HDIM;
  const unsigned short* kp = k + (size_t)tok * DMODEL;
  const unsigned short* vp = v + (size_t)tok * DMODEL;

  const float scale = 0.08838834764831845f;         // 1/sqrt(128)
  float s[8];
  #pragma unroll
  for (int t = 0; t < 8; ++t) {
    float a = 0.f;
    for (int d = 0; d < HDIM; ++d) a += bf2f(qp[d]) * bf2f(kp[t * HDIM + d]);
    s[t] = a * scale;
  }
  float mx = s[0];
  #pragma unroll
  for (int t = 1; t < 8; ++t) mx = fmaxf(mx, s[t]);
  float sum = 0.f;
  #pragma unroll
  for (int t = 0; t < 8; ++t) { s[t] = expf(s[t] - mx); sum += s[t]; }
  const float inv = 1.0f / sum;
  #pragma unroll
  for (int t = 0; t < 8; ++t) {
    s[t] *= inv;
    attnw[(size_t)tok * 64 + h * 8 + t] = s[t];
  }
  for (int d = 0; d < HDIM; ++d) {
    float o = 0.f;
    #pragma unroll
    for (int t = 0; t < 8; ++t) o += s[t] * bf2f(vp[t * HDIM + d]);
    outb[(size_t)tok * DMODEL + h * HDIM + d] = f2bf(o);
  }
}

// ---------------------------------------------------------------------------
// Launch: cvt X -> 5x transpose W -> Q,K,V GEMMs -> attention -> FFN1 -> FFN2
// ---------------------------------------------------------------------------
extern "C" void kernel_launch(void* const* d_in, const int* in_sizes, int n_in,
                              void* d_out, int out_size, void* d_ws, size_t ws_size,
                              hipStream_t stream) {
  const float* x  = (const float*)d_in[0];
  const float* Wq = (const float*)d_in[1];  const float* bq = (const float*)d_in[2];
  const float* Wk = (const float*)d_in[3];  const float* bk = (const float*)d_in[4];
  const float* Wv = (const float*)d_in[5];  const float* bv = (const float*)d_in[6];
  const float* W1 = (const float*)d_in[7];  const float* b1 = (const float*)d_in[8];
  const float* W2 = (const float*)d_in[9];  const float* b2 = (const float*)d_in[10];

  float* out_ff   = (float*)d_out;                                 // [16384*1024]
  float* out_attn = (float*)d_out + (size_t)TOKENS * DMODEL;       // [16384*64]

  // workspace layout (bf16 elements)
  unsigned short* ws = (unsigned short*)d_ws;
  const size_t ACT = (size_t)TOKENS * DMODEL;   // 16,777,216
  const size_t WSZ = (size_t)DMODEL * DMODEL;   //  1,048,576
  size_t o = 0;
  unsigned short* Xb  = ws + o; o += ACT;
  unsigned short* Wtq = ws + o; o += WSZ;
  unsigned short* Wtk = ws + o; o += WSZ;
  unsigned short* Wtv = ws + o; o += WSZ;
  unsigned short* Wt1 = ws + o; o += WSZ;
  unsigned short* Wt2 = ws + o; o += WSZ;
  unsigned short* qb  = ws + o; o += ACT;
  unsigned short* kb  = ws + o; o += ACT;
  unsigned short* vb  = ws + o; o += ACT;
  unsigned short* ab  = ws + o; o += ACT;   // attention output
  unsigned short* f1b = ws + o; o += ACT;   // FFN hidden

  // 1) input -> bf16
  cvt_f32_bf16_kernel<<<4096, 256, 0, stream>>>(x, Xb, (long)(ACT / 4));

  // 2) weights -> [N][K] bf16
  {
    dim3 g(32, 32), b(256);
    transpose_cvt_kernel<<<g, b, 0, stream>>>(Wq, Wtq);
    transpose_cvt_kernel<<<g, b, 0, stream>>>(Wk, Wtk);
    transpose_cvt_kernel<<<g, b, 0, stream>>>(Wv, Wtv);
    transpose_cvt_kernel<<<g, b, 0, stream>>>(W1, Wt1);
    transpose_cvt_kernel<<<g, b, 0, stream>>>(W2, Wt2);
  }

  // 3) GEMMs + attention
  dim3 gg(DMODEL / BN, TOKENS / BM);   // (8, 64)
  gemm_bf16_wmma_kernel<<<gg, 256, 0, stream>>>(Xb, Wtq, bq, qb, 0, 0);
  gemm_bf16_wmma_kernel<<<gg, 256, 0, stream>>>(Xb, Wtk, bk, kb, 0, 0);
  gemm_bf16_wmma_kernel<<<gg, 256, 0, stream>>>(Xb, Wtv, bv, vb, 0, 0);

  attention_kernel<<<(TOKENS * NHEADS) / 256, 256, 0, stream>>>(qb, kb, vb, ab, out_attn);

  gemm_bf16_wmma_kernel<<<gg, 256, 0, stream>>>(ab,  Wt1, b1, f1b,   0, 1);  // ReLU
  gemm_bf16_wmma_kernel<<<gg, 256, 0, stream>>>(f1b, Wt2, b2, out_ff, 1, 0); // fp32 out
}
